// MultiHeadAttention_11570641896178
// MI455X (gfx1250) — matrix operand order repaired and checked by reading, not compiled, not run on hardware
//
#include <hip/hip_runtime.h>
#include <hip/hip_bf16.h>

typedef __attribute__((ext_vector_type(16))) _Float16 v16h;
typedef __attribute__((ext_vector_type(8)))  _Float16 v8h;
typedef __attribute__((ext_vector_type(8)))  float    v8f;
typedef __attribute__((ext_vector_type(4)))  float    v4f;

#define BB    2
#define SS    2048
#define DD    1024
#define HH    16
#define DEPTH 64

// ---------------------------------------------------------------------------
// Wave-level GEMM core. One wave (32 lanes) computes a 16 x (16*NBLK) tile of
//   C = A @ Bt^T   A: row-major [M x K] (f32 or f16), Bt: row-major [N x K] f16
// v_wmma_f32_16x16x32_f16, f32 accumulation. Fragment layouts per ISA 7.12.2.
// ---------------------------------------------------------------------------
template <typename AT, int NBLK>
__device__ __forceinline__ void wave_gemm(
    const AT* __restrict__ A, int lda,
    const _Float16* __restrict__ Bt, int ldb,
    int K, int mtile, int ntile0, v8f acc[NBLK])
{
  const int lane = threadIdx.x & 31;
  const int l15  = lane & 15;
  const int hi   = lane >> 4;
  const AT* arow = A + (size_t)(mtile * 16 + l15) * lda + hi * 8;

  for (int kk = 0; kk < K; kk += 32) {
    v16h a;
    if constexpr (sizeof(AT) == 2) {
      v8h c0 = *(const v8h*)(arow + kk);
      v8h c1 = *(const v8h*)(arow + kk + 16);
      #pragma unroll
      for (int i = 0; i < 8; ++i) { a[i] = c0[i]; a[i + 8] = c1[i]; }
    } else {
      v4f f0 = *(const v4f*)(arow + kk);
      v4f f1 = *(const v4f*)(arow + kk + 4);
      v4f f2 = *(const v4f*)(arow + kk + 16);
      v4f f3 = *(const v4f*)(arow + kk + 20);
      #pragma unroll
      for (int i = 0; i < 4; ++i) {
        a[i]      = (_Float16)f0[i];
        a[i + 4]  = (_Float16)f1[i];
        a[i + 8]  = (_Float16)f2[i];
        a[i + 12] = (_Float16)f3[i];
      }
    }
    #pragma unroll
    for (int j = 0; j < NBLK; ++j) {
      const _Float16* bp =
          Bt + (size_t)((ntile0 + j) * 16 + l15) * ldb + kk + hi * 8;
      v8h b0 = *(const v8h*)bp;
      v8h b1 = *(const v8h*)(bp + 16);
      v16h b;
      #pragma unroll
      for (int i = 0; i < 8; ++i) { b[i] = b0[i]; b[i + 8] = b1[i]; }
      acc[j] = __builtin_amdgcn_wmma_f32_16x16x32_f16(
          false, a, false, b, (short)0, acc[j], false, false);
    }
  }
}

// ---------------------------------------------------------------------------
// Mask bit-pack: one bit per mask element (wave32 ballot). 16.8MB -> 512KB.
// ---------------------------------------------------------------------------
__global__ void k_pack_mask(const int* __restrict__ mask,
                            unsigned* __restrict__ bits)
{
  const int i = blockIdx.x * 256 + threadIdx.x;
  const unsigned long long b = __ballot(mask[i] & 1);
  if ((threadIdx.x & 31) == 0) bits[i >> 5] = (unsigned)b;
}

// ---------------------------------------------------------------------------
// Weight transpose + f16 convert:  W f32 [K][N] -> Wt f16 [N][K]
// ---------------------------------------------------------------------------
__global__ void k_transpose_w(const float* __restrict__ W,
                              _Float16* __restrict__ Wt)
{
  int i = blockIdx.x * blockDim.x + threadIdx.x;
  int k = i >> 10;
  int n = i & (DD - 1);
  Wt[(size_t)n * DD + k] = (_Float16)W[i];
}

// ---------------------------------------------------------------------------
// Projection: out = X @ W + b (f16 head-split).
//   vt_mode==0: out[b][h][s][d]   (Q, K)
//   vt_mode==1: out[b][h][d][s]   (V transposed = Bt layout for ctx GEMM)
// ---------------------------------------------------------------------------
__global__ void k_proj(const float* __restrict__ X,
                       const _Float16* __restrict__ Wt,
                       const float* __restrict__ bias,
                       _Float16* __restrict__ out, int vt_mode)
{
  const int w    = threadIdx.x >> 5;
  const int lane = threadIdx.x & 31;
  const int l15  = lane & 15, hi = lane >> 4;
  const int mtile  = blockIdx.x * 4 + w;
  const int ntile0 = blockIdx.y * 4;

  v8f acc[4] = {};
  wave_gemm<float, 4>(X, DD, Wt, DD, DD, mtile, ntile0, acc);

  #pragma unroll
  for (int j = 0; j < 4; ++j) {
    const int col = (ntile0 + j) * 16 + l15;
    const float bv = bias[col];
    const int h = col >> 6, d = col & (DEPTH - 1);
    #pragma unroll
    for (int r = 0; r < 8; ++r) {
      const int row = mtile * 16 + r + 8 * hi;
      const int b = row >> 11, s = row & (SS - 1);
      const float val = acc[j][r] + bv;
      size_t idx;
      if (vt_mode)
        idx = ((size_t)(b * HH + h) * DEPTH + d) * SS + s;
      else
        idx = ((size_t)(b * HH + h) * SS + s) * DEPTH + d;
      out[idx] = (_Float16)val;
    }
  }
}

// ---------------------------------------------------------------------------
// Fused attention: per wave, 32 q-rows (two 16-row groups) x all 2048 keys of
// one (b,h) slice. Logits computed TRANSPOSED (S^T = kh @ qh^T): lane = q,
// VGPR r = key. kh A-fragments and vt B-fragments are loaded once per 32-key
// block and SHARED by both q-groups (halves L2 traffic). Mask comes from the
// bit-packed table: one u32 per lane per 32-key block.
// Pass 1: online max+sum. Pass 2: recompute, write normalized attn
// (non-temporal, write-once) and accumulate ctx = p @ vh with WMMA.
// ---------------------------------------------------------------------------
__global__ void k_attn_fused(const _Float16* __restrict__ qh,
                             const _Float16* __restrict__ kh,
                             const _Float16* __restrict__ vt,
                             const unsigned* __restrict__ mbits,
                             float* __restrict__ attn,
                             _Float16* __restrict__ concat)
{
  const int wv   = threadIdx.x >> 5;
  const int lane = threadIdx.x & 31;
  const int l15  = lane & 15, hi = lane >> 4;
  const int z  = blockIdx.z;
  const int q0 = (blockIdx.x * 4 + wv) * 32;

  const _Float16* qz = qh + (size_t)z * SS * DEPTH;
  const _Float16* kz = kh + (size_t)z * SS * DEPTH;
  const _Float16* vz = vt + (size_t)z * DEPTH * SS;
  float* az = attn + (size_t)z * SS * SS;

  const unsigned* mrow0 = mbits + (size_t)(q0 + l15) * (SS / 32);
  const unsigned* mrow1 = mbits + (size_t)(q0 + 16 + l15) * (SS / 32);

  // loop-invariant qh B-fragments, two q-groups x two depth k-iters
  const _Float16* qr0 = qz + (size_t)(q0 + l15) * DEPTH + hi * 16;
  const _Float16* qr1 = qz + (size_t)(q0 + 16 + l15) * DEPTH + hi * 16;
  const v16h bq00 = *(const v16h*)(qr0);
  const v16h bq01 = *(const v16h*)(qr0 + 32);
  const v16h bq10 = *(const v16h*)(qr1);
  const v16h bq11 = *(const v16h*)(qr1 + 32);

  // kh A-fragments for one 16-key tile (both depth k-iters)
  auto load_kfrag = [&](int t, v16h& a0, v16h& a1) {
    const _Float16* krow = kz + (size_t)(t * 16 + l15) * DEPTH + hi * 16;
    v8h c0 = *(const v8h*)(krow);
    v8h c1 = *(const v8h*)(krow + 8);
    v8h c2 = *(const v8h*)(krow + 32);
    v8h c3 = *(const v8h*)(krow + 40);
    #pragma unroll
    for (int i = 0; i < 8; ++i) {
      a0[i] = c0[i]; a0[i + 8] = c1[i];
      a1[i] = c2[i]; a1[i + 8] = c3[i];
    }
  };

  // masked+scaled logits for a 32-key pair of tiles, one q-group
  auto pair_logits = [&](const v16h& x0, const v16h& x1,
                         const v16h& y0, const v16h& y1,
                         const v16h& b0, const v16h& b1,
                         unsigned wm, v8f& s0, v8f& s1) {
    v8f d0 = {}, d1 = {};
    d0 = __builtin_amdgcn_wmma_f32_16x16x32_f16(false, x0, false, b0,
                                                (short)0, d0, false, false);
    d0 = __builtin_amdgcn_wmma_f32_16x16x32_f16(false, x1, false, b1,
                                                (short)0, d0, false, false);
    d1 = __builtin_amdgcn_wmma_f32_16x16x32_f16(false, y0, false, b0,
                                                (short)0, d1, false, false);
    d1 = __builtin_amdgcn_wmma_f32_16x16x32_f16(false, y1, false, b1,
                                                (short)0, d1, false, false);
    const unsigned e0 = (wm >> (hi * 8)) & 0xffu;        // keys 32t2+8hi+r
    const unsigned e1 = (wm >> (16 + hi * 8)) & 0xffu;   // keys 32t2+16+8hi+r
    #pragma unroll
    for (int r = 0; r < 8; ++r) {
      s0[r] = d0[r] * 0.125f - 1e9f * (float)((e0 >> r) & 1u);
      s1[r] = d1[r] * 0.125f - 1e9f * (float)((e1 >> r) & 1u);
    }
  };

  // online softmax state update over a 32-key pair
  auto upd = [&](float& m, float& l, const v8f& s0, const v8f& s1) {
    float tm = fmaxf(s0[0], s1[0]);
    #pragma unroll
    for (int r = 1; r < 8; ++r) tm = fmaxf(tm, fmaxf(s0[r], s1[r]));
    const float mn = fmaxf(m, tm);
    float ts = 0.0f;
    #pragma unroll
    for (int r = 0; r < 8; ++r)
      ts += __expf(s0[r] - mn) + __expf(s1[r] - mn);
    l = l * __expf(m - mn) + ts;
    m = mn;
  };

  // ---- pass 1 ----
  float mA = -3.0e38f, lA = 0.0f, mB = -3.0e38f, lB = 0.0f;
  for (int t2 = 0; t2 < SS / 32; ++t2) {
    v16h x0, x1, y0, y1;
    load_kfrag(2 * t2, x0, x1);
    load_kfrag(2 * t2 + 1, y0, y1);
    const unsigned w0 = mrow0[t2], w1 = mrow1[t2];
    v8f s0, s1;
    pair_logits(x0, x1, y0, y1, bq00, bq01, w0, s0, s1);
    upd(mA, lA, s0, s1);
    pair_logits(x0, x1, y0, y1, bq10, bq11, w1, s0, s1);
    upd(mB, lB, s0, s1);
  }
  // merge the two half-wave key groups (lane pairs l, l^16)
  auto merge = [&](float m, float l, float& mf, float& invl) {
    const float mo = __shfl_xor(m, 16);
    const float lo = __shfl_xor(l, 16);
    mf = fmaxf(m, mo);
    const float lf = l * __expf(m - mf) + lo * __expf(mo - mf);
    invl = 1.0f / lf;
  };
  float mfA, invA, mfB, invB;
  merge(mA, lA, mfA, invA);
  merge(mB, lB, mfB, invB);

  // ---- pass 2 ----
  v8f oA[4] = {}, oB[4] = {};
  for (int t2 = 0; t2 < SS / 32; ++t2) {
    v16h x0, x1, y0, y1;
    load_kfrag(2 * t2, x0, x1);
    load_kfrag(2 * t2 + 1, y0, y1);
    const unsigned w0 = mrow0[t2], w1 = mrow1[t2];

    auto emit = [&](const v8f& s0, const v8f& s1, float mf, float invl,
                    int qrow) -> v16h {
      v16h pa;
      v4f e0, e1, e2, e3;
      #pragma unroll
      for (int r = 0; r < 8; ++r) {
        const float p0 = __expf(s0[r] - mf) * invl;
        const float p1 = __expf(s1[r] - mf) * invl;
        pa[r]     = (_Float16)p0;
        pa[r + 8] = (_Float16)p1;
        if (r < 4) { e0[r] = p0; e2[r] = p1; }
        else       { e1[r - 4] = p0; e3[r - 4] = p1; }
      }
      float* wp = az + (size_t)qrow * SS + t2 * 32 + hi * 8;
      __builtin_nontemporal_store(e0, (v4f*)(wp));
      __builtin_nontemporal_store(e1, (v4f*)(wp + 4));
      __builtin_nontemporal_store(e2, (v4f*)(wp + 16));
      __builtin_nontemporal_store(e3, (v4f*)(wp + 20));
      return pa;
    };

    v8f s0, s1;
    pair_logits(x0, x1, y0, y1, bq00, bq01, w0, s0, s1);
    v16h paA = emit(s0, s1, mfA, invA, q0 + l15);
    pair_logits(x0, x1, y0, y1, bq10, bq11, w1, s0, s1);
    v16h paB = emit(s0, s1, mfB, invB, q0 + 16 + l15);

    // ctx += p @ vh over this 32-key block (vt B-frags shared by both groups)
    #pragma unroll
    for (int j = 0; j < 4; ++j) {
      const _Float16* bp = vz + (size_t)(j * 16 + l15) * SS + t2 * 32 + hi * 8;
      v8h bv0 = *(const v8h*)bp;
      v8h bv1 = *(const v8h*)(bp + 16);
      v16h bv;
      #pragma unroll
      for (int i = 0; i < 8; ++i) { bv[i] = bv0[i]; bv[i + 8] = bv1[i]; }
      oA[j] = __builtin_amdgcn_wmma_f32_16x16x32_f16(false, paA, false, bv,
                                                     (short)0, oA[j], false, false);
      oB[j] = __builtin_amdgcn_wmma_f32_16x16x32_f16(false, paB, false, bv,
                                                     (short)0, oB[j], false, false);
    }
  }

  // epilogue: ctx -> concat f16 [B*S][D]
  const int b = z >> 4, h = z & (HH - 1);
  #pragma unroll
  for (int j = 0; j < 4; ++j) {
    const int d = j * 16 + l15;
    #pragma unroll
    for (int r = 0; r < 8; ++r) {
      const int sA = q0 + r + 8 * hi;
      const int sB = q0 + 16 + r + 8 * hi;
      concat[(size_t)(b * SS + sA) * DD + h * DEPTH + d] = (_Float16)oA[j][r];
      concat[(size_t)(b * SS + sB) * DD + h * DEPTH + d] = (_Float16)oB[j][r];
    }
  }
}

// ---------------------------------------------------------------------------
// Dense: out = A @ Wo + bo  (A f16 [B*S][D], Wt f16 [D][D], out f16 or f32)
// ---------------------------------------------------------------------------
template <typename OT>
__global__ void k_dense(const _Float16* __restrict__ A,
                        const _Float16* __restrict__ Wt,
                        const float* __restrict__ bias,
                        OT* __restrict__ out)
{
  const int w    = threadIdx.x >> 5;
  const int lane = threadIdx.x & 31;
  const int l15  = lane & 15, hi = lane >> 4;
  const int mtile  = blockIdx.x * 4 + w;
  const int ntile0 = blockIdx.y * 4;

  v8f acc[4] = {};
  wave_gemm<_Float16, 4>(A, DD, Wt, DD, DD, mtile, ntile0, acc);

  #pragma unroll
  for (int j = 0; j < 4; ++j) {
    const int col = (ntile0 + j) * 16 + l15;
    const float bv = bias[col];
    #pragma unroll
    for (int r = 0; r < 8; ++r) {
      const int row = mtile * 16 + r + 8 * hi;
      out[(size_t)row * DD + col] = (OT)(acc[j][r] + bv);
    }
  }
}

// ---------------------------------------------------------------------------
extern "C" void kernel_launch(void* const* d_in, const int* in_sizes, int n_in,
                              void* d_out, int out_size, void* d_ws,
                              size_t ws_size, hipStream_t stream)
{
  (void)in_sizes; (void)n_in; (void)out_size; (void)ws_size;

  const float* v   = (const float*)d_in[0];
  const float* k   = (const float*)d_in[1];
  const float* q   = (const float*)d_in[2];
  const float* Wq  = (const float*)d_in[3];
  const float* bq  = (const float*)d_in[4];
  const float* Wk  = (const float*)d_in[5];
  const float* bk  = (const float*)d_in[6];
  const float* Wv  = (const float*)d_in[7];
  const float* bv  = (const float*)d_in[8];
  const float* Wo  = (const float*)d_in[9];
  const float* bo  = (const float*)d_in[10];
  const int*  mask = (const int*)d_in[11];

  float* out  = (float*)d_out;
  float* attn = out + (size_t)BB * SS * DD;  // tuple: (out, attn) flat

  // workspace layout (~49 MB)
  char* ws = (char*)d_ws;
  size_t off = 0;
  auto alloc = [&](size_t bytes) -> void* {
    void* p = ws + off;
    off += (bytes + 255) & ~(size_t)255;
    return p;
  };
  _Float16* qh     = (_Float16*)alloc((size_t)BB * SS * DD * 2);
  _Float16* kh     = (_Float16*)alloc((size_t)BB * SS * DD * 2);
  _Float16* vt     = (_Float16*)alloc((size_t)BB * SS * DD * 2);
  _Float16* wqt    = (_Float16*)alloc((size_t)DD * DD * 2);
  _Float16* wkt    = (_Float16*)alloc((size_t)DD * DD * 2);
  _Float16* wvt    = (_Float16*)alloc((size_t)DD * DD * 2);
  _Float16* wot    = (_Float16*)alloc((size_t)DD * DD * 2);
  _Float16* concat = (_Float16*)alloc((size_t)BB * SS * DD * 2);
  _Float16* out1   = (_Float16*)alloc((size_t)BB * SS * DD * 2);
  unsigned* mbits  = (unsigned*)alloc((size_t)SS * SS / 32 * 4);

  const dim3 blk(128);  // 4 waves

  // 1) mask bit-pack + weight transpose/convert
  k_pack_mask<<<SS * SS / 256, 256, 0, stream>>>(mask, mbits);
  k_transpose_w<<<DD * DD / 256, 256, 0, stream>>>(Wq, wqt);
  k_transpose_w<<<DD * DD / 256, 256, 0, stream>>>(Wk, wkt);
  k_transpose_w<<<DD * DD / 256, 256, 0, stream>>>(Wv, wvt);
  k_transpose_w<<<DD * DD / 256, 256, 0, stream>>>(Wo, wot);

  // 2) projections
  const dim3 gproj(BB * SS / 64, DD / 64);  // (64,16)
  k_proj<<<gproj, blk, 0, stream>>>(q, wqt, bq, qh, 0);
  k_proj<<<gproj, blk, 0, stream>>>(k, wkt, bk, kh, 0);
  k_proj<<<gproj, blk, 0, stream>>>(v, wvt, bv, vt, 1);

  // 3) fused logits + softmax + attn-write + ctx (wave = 32 q rows)
  const dim3 gat(SS / 128, 1, BB * HH);  // (16,1,32)
  k_attn_fused<<<gat, blk, 0, stream>>>(qh, kh, vt, mbits, attn, concat);

  // 4) two chained output denses (reference applies Wo twice)
  const dim3 gd(BB * SS / 64, DD / 64);
  k_dense<_Float16><<<gd, blk, 0, stream>>>(concat, wot, bo, out1);
  k_dense<float><<<gd, blk, 0, stream>>>(out1, wot, bo, out);
}
